// NaiveLSTM_37632503447561
// MI455X (gfx1250) — matrix operand order repaired and checked by reading, not compiled, not run on hardware
//
#include <hip/hip_runtime.h>
#include <hip/hip_bf16.h>

typedef __attribute__((ext_vector_type(16))) _Float16 v16h;
typedef __attribute__((ext_vector_type(8)))  float    v8f;
typedef __attribute__((ext_vector_type(4)))  unsigned int v4u;

#define BB 64
#define SS 512
#define ID 1024
#define HD 1024
#define KD 2048   // I + H
#define N4 4096   // 4*H

union FragU { v16h v; v4u u[2]; };

__device__ __forceinline__ float sigmoidf_(float x) { return 1.0f / (1.0f + __expf(-x)); }

// One-time: build Wcat f16 [4096 x 2048] (rows = gate outputs i,f,g,o; cols = [x | h])
// and fused bias f32 [4096] (b_i* + b_h*).
__global__ void prep_weights(
    const float* __restrict__ w_ii, const float* __restrict__ w_hi,
    const float* __restrict__ w_if, const float* __restrict__ w_hf,
    const float* __restrict__ w_ig, const float* __restrict__ w_hg,
    const float* __restrict__ w_io, const float* __restrict__ w_ho,
    const float* __restrict__ b_ii, const float* __restrict__ b_hi,
    const float* __restrict__ b_if, const float* __restrict__ b_hf,
    const float* __restrict__ b_ig, const float* __restrict__ b_hg,
    const float* __restrict__ b_io, const float* __restrict__ b_ho,
    _Float16* __restrict__ Wcat, float* __restrict__ bias)
{
    size_t idx = (size_t)blockIdx.x * blockDim.x + threadIdx.x;
    if (idx >= (size_t)N4 * KD) return;
    int n = (int)(idx / KD), k = (int)(idx % KD);
    int g = n >> 10, r = n & 1023;
    const float* Wi = (g == 0) ? w_ii : (g == 1) ? w_if : (g == 2) ? w_ig : w_io;
    const float* Wh = (g == 0) ? w_hi : (g == 1) ? w_hf : (g == 2) ? w_hg : w_ho;
    float v = (k < ID) ? Wi[(size_t)r * ID + k] : Wh[(size_t)r * HD + (k - ID)];
    Wcat[idx] = (_Float16)v;
    if (k == 0) {
        const float* bi = (g == 0) ? b_ii : (g == 1) ? b_if : (g == 2) ? b_ig : b_io;
        const float* bh = (g == 0) ? b_hi : (g == 1) ? b_hf : (g == 2) ? b_hg : b_ho;
        bias[n] = bi[r] + bh[r];
    }
}

// Init: c = c0, xh0 = [f16(x[:,0,:]) | f16(h0)]
__global__ void init_state(const float* __restrict__ x,
                           const float* __restrict__ h0,
                           const float* __restrict__ c0,
                           _Float16* __restrict__ xh0,
                           float* __restrict__ c_ws)
{
    int idx = blockIdx.x * blockDim.x + threadIdx.x; // 64*1024
    int b = idx >> 10, h = idx & 1023;
    c_ws[idx] = c0[idx];
    xh0[(size_t)b * KD + ID + h] = (_Float16)h0[idx];
    xh0[(size_t)b * KD + h]      = (_Float16)x[(size_t)b * SS * ID + h]; // t = 0
}

// One LSTM timestep. 64 blocks x 512 threads (16 wave32).
// Block = 16 h-columns. Wave (g, mt) computes the 16x16 WMMA tile for gate g,
// batch rows [mt*16, mt*16+16), K = 2048. Tiles -> LDS -> cell update.
__global__ __launch_bounds__(512) void lstm_step(
    const float*   __restrict__ x,        // [B,S,I]
    const _Float16* __restrict__ Wcat,    // [4096,2048]
    const float*   __restrict__ bias,     // [4096]
    const _Float16* __restrict__ xh_cur,  // [B,2048]  (read)
    _Float16*      __restrict__ xh_next,  // [B,2048]  (write, ping-pong)
    float*         __restrict__ c_ws,     // [B,H]
    float*         __restrict__ out,      // hidden_seq [B,S,H] ++ h_f ++ c_f
    int t)
{
    __shared__ float gates[4 * 64 * 16]; // [gate][m=batch][col] = 16 KB

    const int tid  = threadIdx.x;
    const int lane = tid & 31;
    const int wave = tid >> 5;       // 0..15
    const int g    = wave & 3;       // gate
    const int mt   = wave >> 2;      // M tile (batch/16)
    const int n0   = blockIdx.x << 4;// h-column base

    const int col = lane & 15;
    const int hi  = lane >> 4;
    const int nrow = (g << 10) + n0 + col;   // Wcat row (N index)
    const int brow = (mt << 4) + col;        // batch row (M index)

    float bv = bias[nrow];
    v8f acc;
#pragma unroll
    for (int r = 0; r < 8; ++r) acc[r] = bv;

    // A: 16-bit 16x32 layout — lane holds K = {hi*8 .. hi*8+7} and {16+hi*8 ..}
    // B: 16-bit 32x16 layout — same K order as A: {hi*8 .. hi*8+7} and {16+hi*8 ..}
    const _Float16* ap = xh_cur + (size_t)brow * KD + (hi << 3);
    const _Float16* bp = Wcat   + (size_t)nrow * KD + (hi << 3);

#pragma unroll 4
    for (int k0 = 0; k0 < KD; k0 += 32) {
        FragU a, bfr;
        a.u[0]   = *(const v4u*)(ap + k0);
        a.u[1]   = *(const v4u*)(ap + k0 + 16);
        bfr.u[0] = *(const v4u*)(bp + k0);
        bfr.u[1] = *(const v4u*)(bp + k0 + 16);
        acc = __builtin_amdgcn_wmma_f32_16x16x32_f16(
            /*neg_a=*/false, a.v, /*neg_b=*/false, bfr.v,
            /*c_mod=*/(short)0, acc, /*reuse_a=*/false, /*reuse_b=*/false);
    }

    // C/D layout: lane col = lane%16; VGPR r holds M = r + (lane>=16 ? 8 : 0)
#pragma unroll
    for (int r = 0; r < 8; ++r) {
        int m = (mt << 4) + (hi << 3) + r;
        gates[((g << 6) + m) * 16 + col] = acc[r];
    }
    __syncthreads();

    // ---- cell update for this block's 16 h-columns, all 64 batches ----
    for (int e = tid; e < 1024; e += 512) {
        int b  = e >> 4;
        int c4 = e & 15;
        int h  = n0 + c4;
        float gi = gates[(0 * 64 + b) * 16 + c4];
        float gf = gates[(1 * 64 + b) * 16 + c4];
        float gg = gates[(2 * 64 + b) * 16 + c4];
        float go = gates[(3 * 64 + b) * 16 + c4];
        float i_ = sigmoidf_(gi);
        float f_ = sigmoidf_(gf);
        float g_ = tanhf(gg);
        float o_ = sigmoidf_(go);
        size_t ch = (size_t)b * HD + h;
        float cn = f_ * c_ws[ch] + i_ * g_;
        c_ws[ch] = cn;
        float hn = o_ * tanhf(cn);
        out[((size_t)b * SS + t) * HD + h] = hn;
        xh_next[(size_t)b * KD + ID + h] = (_Float16)hn;
        if (t + 1 < SS) {
            xh_next[(size_t)b * KD + h] =
                (_Float16)x[((size_t)b * SS + (t + 1)) * ID + h];
        } else {
            // final h / c outputs
            out[(size_t)BB * SS * HD + (size_t)b * HD + h] = hn;
            out[(size_t)BB * SS * HD + (size_t)BB * HD + (size_t)b * HD + h] = cn;
        }
    }
}

extern "C" void kernel_launch(void* const* d_in, const int* in_sizes, int n_in,
                              void* d_out, int out_size, void* d_ws, size_t ws_size,
                              hipStream_t stream) {
    const float* x    = (const float*)d_in[0];
    const float* h0   = (const float*)d_in[1];
    const float* c0   = (const float*)d_in[2];
    const float* w_ii = (const float*)d_in[3];
    const float* w_hi = (const float*)d_in[4];
    const float* b_ii = (const float*)d_in[5];
    const float* b_hi = (const float*)d_in[6];
    const float* w_if = (const float*)d_in[7];
    const float* w_hf = (const float*)d_in[8];
    const float* b_if = (const float*)d_in[9];
    const float* b_hf = (const float*)d_in[10];
    const float* w_ig = (const float*)d_in[11];
    const float* w_hg = (const float*)d_in[12];
    const float* b_ig = (const float*)d_in[13];
    const float* b_hg = (const float*)d_in[14];
    const float* w_io = (const float*)d_in[15];
    const float* w_ho = (const float*)d_in[16];
    const float* b_io = (const float*)d_in[17];
    const float* b_ho = (const float*)d_in[18];

    char* ws = (char*)d_ws;
    size_t off = 0;
    _Float16* Wcat = (_Float16*)(ws + off); off += (size_t)N4 * KD * sizeof(_Float16); // 16 MB
    float*    bias = (float*)(ws + off);    off += (size_t)N4 * sizeof(float);
    _Float16* xh0  = (_Float16*)(ws + off); off += (size_t)BB * KD * sizeof(_Float16);
    _Float16* xh1  = (_Float16*)(ws + off); off += (size_t)BB * KD * sizeof(_Float16);
    float*    c_ws = (float*)(ws + off);    off += (size_t)BB * HD * sizeof(float);

    {
        size_t total = (size_t)N4 * KD;
        int blocks = (int)((total + 255) / 256);
        prep_weights<<<blocks, 256, 0, stream>>>(
            w_ii, w_hi, w_if, w_hf, w_ig, w_hg, w_io, w_ho,
            b_ii, b_hi, b_if, b_hf, b_ig, b_hg, b_io, b_ho,
            Wcat, bias);
    }
    init_state<<<(BB * HD) / 256, 256, 0, stream>>>(x, h0, c0, xh0, c_ws);

    for (int t = 0; t < SS; ++t) {
        const _Float16* cur = (t & 1) ? xh1 : xh0;
        _Float16*       nxt = (t & 1) ? xh0 : xh1;
        lstm_step<<<64, 512, 0, stream>>>(x, Wcat, bias, cur, nxt, c_ws,
                                          (float*)d_out, t);
    }
}
